// Transformer_LM_9878424781062
// MI455X (gfx1250) — matrix operand order repaired and checked by reading, not compiled, not run on hardware
//
#include <hip/hip_runtime.h>

// Transformer LM forward on gfx1250 (MI455X), wave32 + WMMA f16->f32.
// L=4, D=1024, NH=16, DH=64, V=32000, B=2, T=2048.
// Data movement: TDM (tensor_load_to_lds) for GEMM A tiles (double-buffered),
// global_load_async_to_lds_b128 for attention Q/K tiles.

#define L_  4
#define D_  1024
#define NH_ 16
#define DH_ 64
#define V_  32000
#define B_  2
#define T_  2048
#define M_  (B_*T_)   // 4096 token rows

typedef __attribute__((ext_vector_type(16))) _Float16     v16h;
typedef __attribute__((ext_vector_type(8)))  _Float16     v8h;
typedef __attribute__((ext_vector_type(8)))  float        v8f;
typedef __attribute__((ext_vector_type(4)))  unsigned int u32x4;
typedef __attribute__((ext_vector_type(8)))  int          i32x8;
typedef __attribute__((ext_vector_type(4)))  int          i32x4;

#define FLAG_BIAS   1
#define FLAG_SELU   2
#define FLAG_OUTF32 4

#if __has_builtin(__builtin_amdgcn_tensor_load_to_lds)
#define HAVE_TDM 1
#else
#define HAVE_TDM 0
#endif

// Low 32 bits of a flat pointer into the LDS aperture == LDS byte offset
// (ISA 10.2).
__device__ __forceinline__ unsigned int lds_of(const void* p) {
  return (unsigned int)(unsigned long long)(uintptr_t)p;
}

// Per-lane async copy of 16 bytes global -> LDS (ASYNCcnt-tracked).
__device__ __forceinline__ void async_b128(unsigned int lds_off, const void* g) {
  asm volatile("global_load_async_to_lds_b128 %0, %1, off"
               :: "v"(lds_off), "v"((unsigned long long)(uintptr_t)g)
               : "memory");
}
__device__ __forceinline__ void wait_async_le0() {
  asm volatile("s_wait_asynccnt 0x0" ::: "memory");
}
__device__ __forceinline__ void wait_async_le2() {
  asm volatile("s_wait_asynccnt 0x2" ::: "memory");
}

#if HAVE_TDM
// TDM: 2D tile of 16-bit elements, tile0 x tile1 (row-major, row stride
// stride0 elements) -> contiguous LDS at lds_off. One DMA per call.
__device__ __forceinline__ void tdm_load_2d(unsigned int lds_off,
                                            const void* gptr,
                                            unsigned int tile0,
                                            unsigned int tile1,
                                            unsigned long long stride0) {
  unsigned long long ga = (unsigned long long)(uintptr_t)gptr;
  u32x4 g0;
  g0[0] = 1u;                                      // count=1, is_restore=0
  g0[1] = lds_off;                                 // lds_addr
  g0[2] = (unsigned int)ga;                        // global_addr[31:0]
  g0[3] = ((unsigned int)(ga >> 32) & 0x01FFFFFFu) // global_addr[56:32]
        | 0x80000000u;                             // type=2 (bits 127:126)
  const unsigned int td0 = 0x7FFFFFFFu, td1 = 0x7FFFFFFFu; // in-bounds tiles
  i32x8 g1;
  g1[0] = (int)(1u << 16);                         // data_size = 2 bytes
  g1[1] = (int)((td0 & 0xFFFFu) << 16);            // tensor_dim0[15:0]
  g1[2] = (int)(((td0 >> 16) & 0xFFFFu) | ((td1 & 0xFFFFu) << 16));
  g1[3] = (int)(((td1 >> 16) & 0xFFFFu) | (tile0 << 16));   // tile_dim0
  g1[4] = (int)(tile1 & 0xFFFFu);                  // tile_dim1 (tile_dim2=0)
  g1[5] = (int)(unsigned int)(stride0 & 0xFFFFFFFFu);       // dim0_stride lo
  g1[6] = (int)(unsigned int)((stride0 >> 32) & 0xFFFFu);   // dim0_stride hi
  g1[7] = 0;
  i32x4 z4 = {0, 0, 0, 0};
#if defined(__clang_major__) && (__clang_major__ >= 23)
  i32x8 z8 = {0, 0, 0, 0, 0, 0, 0, 0};
  __builtin_amdgcn_tensor_load_to_lds(g0, g1, z4, z4, z8, 0);
#else
  __builtin_amdgcn_tensor_load_to_lds(g0, g1, z4, z4, 0);
#endif
}
#endif

__device__ __forceinline__ v16h cat16(v8h a, v8h b) {
  return __builtin_shufflevector(a, b, 0,1,2,3,4,5,6,7,8,9,10,11,12,13,14,15);
}

// A fragment (16x32 f16, MxK): lanes 0-15 M=lane, VGPR0-3=K0..7, VGPR4-7=K16..23;
// lanes 16-31 K8..15 / K24..31 -> two contiguous 16B LDS reads per lane.
__device__ __forceinline__ v16h load_a_frag(const _Float16* base, int ld) {
  int lane = threadIdx.x & 31;
  int row  = lane & 15, half = lane >> 4;
  v8h a = *(const v8h*)(base + row*ld + half*8);
  v8h b = *(const v8h*)(base + row*ld + 16 + half*8);
  return cat16(a, b);
}

// B fragment (32x16 f16, KxN) from K-contiguous (transposed) storage.
__device__ __forceinline__ v16h load_b_frag(const _Float16* base, int ld) {
  int lane = threadIdx.x & 31;
  int col  = lane & 15, half = lane >> 4;
  v8h a = *(const v8h*)(base + col*ld + half*8);
  v8h b = *(const v8h*)(base + col*ld + 16 + half*8);
  return cat16(a, b);
}

#define WMMA_F32_F16(A,B,C) \
  __builtin_amdgcn_wmma_f32_16x16x32_f16(false,(A),false,(B),(short)0,(C),false,false)

// ---------------------------------------------------------------------------
// GEMM: C[M,N] = A[M,K](f16) x B[K,N](f16) (+bias, +SELU), out f16/f32.
// Block tile 128x64, BK=32. 8 waves, each 32x32 (2x2 WMMA frags).
// A tiles double-buffered via TDM (or per-lane async copies as fallback).
// ---------------------------------------------------------------------------
__global__ __launch_bounds__(256) void k_gemm(const _Float16* __restrict__ A,
                                              const _Float16* __restrict__ Bm,
                                              const float* __restrict__ bias,
                                              float* __restrict__ Cf,
                                              _Float16* __restrict__ Ch,
                                              int Mdim, int Ndim, int Kdim,
                                              int flags)
{
  __shared__ _Float16 sA[2*128*32];  // double-buffered [row][k], 16 KB
  __shared__ _Float16 sBt[64*32];    // [n][k] transposed, 4 KB
  const int t    = threadIdx.x;
  const int n0   = blockIdx.x * 64;
  const int m0   = blockIdx.y * 128;
  const int wid  = t >> 5, lane = t & 31;
  const int wm   = wid >> 1, wn = wid & 1;   // 4x2 wave grid
  const int lr   = lane & 15, lh = lane >> 4;
  const int nk   = Kdim >> 5;

#if HAVE_TDM
  // One DMA per tile, issued by wave 0 only (TDM issues once per wave).
  if (wid == 0)
    tdm_load_2d(lds_of(&sA[0]), A + (size_t)m0*Kdim, 32, 128, (size_t)Kdim);
#else
  #pragma unroll
  for (int it = 0; it < 2; ++it) {
    int s = t + it*256;
    int row = s >> 2, seg = s & 3;
    async_b128(lds_of(&sA[row*32 + seg*8]),
               A + (size_t)(m0+row)*Kdim + seg*8);
  }
#endif

  v8f acc00 = {}, acc01 = {}, acc10 = {}, acc11 = {};

  for (int kt = 0; kt < nk; ++kt) {
    const int k0  = kt << 5;
    const int buf = kt & 1;
    __syncthreads();   // prior iteration done reading sBt / writing-buffer safe

    // Stage B tile transposed: read row-major 8-half chunk, scatter columns.
    {
      int k = t >> 3, nb = t & 7;
      v8h q = *(const v8h*)(Bm + (size_t)(k0+k)*Ndim + n0 + nb*8);
      #pragma unroll
      for (int i = 0; i < 8; ++i) sBt[(nb*8+i)*32 + k] = q[i];
    }
    if (kt + 1 < nk)   // global_prefetch_b8 for next B tile
      __builtin_prefetch(Bm + (size_t)(k0 + 32 + (t>>3))*Ndim + n0, 0, 0);

#if HAVE_TDM
    if (wid == 0) {
      if (kt + 1 < nk) {
        tdm_load_2d(lds_of(&sA[(buf^1)*4096]),
                    A + (size_t)m0*Kdim + (k0 + 32), 32, 128, (size_t)Kdim);
        __builtin_amdgcn_s_wait_tensorcnt((short)1);  // tile kt complete
      } else {
        __builtin_amdgcn_s_wait_tensorcnt((short)0);
      }
    }
#else
    if (kt + 1 < nk) {
      #pragma unroll
      for (int it = 0; it < 2; ++it) {
        int s = t + it*256;
        int row = s >> 2, seg = s & 3;
        async_b128(lds_of(&sA[(buf^1)*4096 + row*32 + seg*8]),
                   A + (size_t)(m0+row)*Kdim + (k0 + 32) + seg*8);
      }
      wait_async_le2();   // the older tile's 2 per-thread copies are done
    } else {
      wait_async_le0();
    }
#endif
    __syncthreads();

    const _Float16* aT = &sA[buf*4096];
    v16h a0 = load_a_frag(aT + (wm*32)*32,    32);
    v16h a1 = load_a_frag(aT + (wm*32+16)*32, 32);
    v16h b0 = load_b_frag(&sBt[(wn*32)*32],    32);
    v16h b1 = load_b_frag(&sBt[(wn*32+16)*32], 32);
    acc00 = WMMA_F32_F16(a0, b0, acc00);
    acc01 = WMMA_F32_F16(a0, b1, acc01);
    acc10 = WMMA_F32_F16(a1, b0, acc10);
    acc11 = WMMA_F32_F16(a1, b1, acc11);
  }

  const float lam = 1.0507009873554805f, alp = 1.6732632423543772f;
  v8f accs[2][2] = {{acc00, acc01}, {acc10, acc11}};
  #pragma unroll
  for (int fm = 0; fm < 2; ++fm)
  #pragma unroll
  for (int fn = 0; fn < 2; ++fn) {
    int nn = n0 + wn*32 + fn*16 + lr;
    float bv = (flags & FLAG_BIAS) ? bias[nn] : 0.0f;
    int mb = m0 + wm*32 + fm*16 + lh*8;
    #pragma unroll
    for (int r = 0; r < 8; ++r) {
      float v = accs[fm][fn][r] + bv;
      if (flags & FLAG_SELU) v = (v > 0.0f) ? lam*v : lam*alp*(__expf(v) - 1.0f);
      size_t idx = (size_t)(mb + r)*Ndim + nn;
      if (flags & FLAG_OUTF32) Cf[idx] = v;
      else                     Ch[idx] = (_Float16)v;
    }
  }
}

// ---------------------------------------------------------------------------
// Flash attention: block = (batch*head, 64-query tile). Causal, streaming
// softmax; O accumulators live in VGPRs across the key-tile loop.
// Q/K tiles staged with async global->LDS copies.
// ---------------------------------------------------------------------------
__global__ __launch_bounds__(256) void k_attn(const _Float16* __restrict__ Q,
                                              const _Float16* __restrict__ K,
                                              const _Float16* __restrict__ Vv,
                                              float* __restrict__ Out)
{
  __shared__ _Float16 sQ [64*64];   // [i][d]  8 KB
  __shared__ _Float16 sK [64*64];   // [j][d]  8 KB (K-contig per column of S)
  __shared__ _Float16 sVt[64*64];   // [d][j]  8 KB (K-contig for P*V)
  __shared__ float    sS [64*64];   // 16 KB
  __shared__ _Float16 sP [64*64];   // 8 KB
  __shared__ float sM[64], sL[64], sAl[64];

  const int t  = threadIdx.x;
  const int i0 = blockIdx.x * 64;
  const int bh = blockIdx.y;
  const int b  = bh / NH_, h = bh % NH_;
  const int wid = t >> 5, lane = t & 31;
  const int wm  = wid >> 1, wn = wid & 1;    // waves: 4 in M x 2 in N
  const int lr  = lane & 15, lh = lane >> 4;
  const float scale = 0.125f;                // 1/sqrt(DH)

  // Async-stage Q tile (64x64 halves = 512 v8h slots, 2 per thread).
  #pragma unroll
  for (int it = 0; it < 2; ++it) {
    int s = t + it*256;
    int row = s >> 3, seg = s & 7;
    async_b128(lds_of(&sQ[row*64 + seg*8]),
               Q + ((size_t)(b*T_ + i0 + row))*D_ + h*DH_ + seg*8);
  }
  if (t < 64) { sM[t] = -3.0e38f; sL[t] = 0.0f; }

  v8f o0 = {}, o1 = {};
  for (int j0 = 0; j0 <= i0; j0 += 64) {
    __syncthreads();   // previous iteration done with sK/sVt/sP
    #pragma unroll
    for (int it = 0; it < 2; ++it) {
      int s = t + it*256;
      int row = s >> 3, seg = s & 7;
      async_b128(lds_of(&sK[row*64 + seg*8]),
                 K + ((size_t)(b*T_ + j0 + row))*D_ + h*DH_ + seg*8);
      v8h q = *(const v8h*)(Vv + ((size_t)(b*T_ + j0 + row))*D_ + h*DH_ + seg*8);
      #pragma unroll
      for (int i = 0; i < 8; ++i) sVt[(seg*8+i)*64 + row] = q[i];
    }
    wait_async_le0();
    __syncthreads();

    // S = Q * K^T : each wave computes a 16x32 strip, DH=64 -> 2 k-steps.
    #pragma unroll
    for (int fn = 0; fn < 2; ++fn) {
      v8f s_acc = {};
      #pragma unroll
      for (int kk = 0; kk < 2; ++kk) {
        v16h af = load_a_frag(&sQ[(wm*16)*64 + kk*32], 64);
        v16h bf = load_b_frag(&sK[(wn*32 + fn*16)*64 + kk*32], 64);
        s_acc = WMMA_F32_F16(af, bf, s_acc);
      }
      #pragma unroll
      for (int r = 0; r < 8; ++r)
        sS[(wm*16 + lh*8 + r)*64 + (wn*32 + fn*16 + lr)] = s_acc[r];
    }
    __syncthreads();

    // Streaming softmax (one thread per query row).
    if (t < 64) {
      int rg = i0 + t;
      float mold = sM[t], mmax = mold;
      for (int j = 0; j < 64; ++j) {
        float v = ((j0 + j) <= rg) ? sS[t*64 + j]*scale : -3.0e38f;
        mmax = fmaxf(mmax, v);
      }
      float alpha = __expf(mold - mmax);
      float lsum  = sL[t]*alpha;
      for (int j = 0; j < 64; ++j) {
        float p = ((j0 + j) <= rg) ? __expf(sS[t*64 + j]*scale - mmax) : 0.0f;
        sP[t*64 + j] = (_Float16)p;
        lsum += p;
      }
      sM[t] = mmax; sL[t] = lsum; sAl[t] = alpha;
    }
    __syncthreads();

    // Rescale O accumulators, then O += P * V.
    #pragma unroll
    for (int r = 0; r < 8; ++r) {
      float al = sAl[wm*16 + lh*8 + r];
      o0[r] *= al; o1[r] *= al;
    }
    #pragma unroll
    for (int kk = 0; kk < 2; ++kk) {
      v16h pf = load_a_frag(&sP[(wm*16)*64 + kk*32], 64);
      v16h v0 = load_b_frag(&sVt[(wn*32 +  0)*64 + kk*32], 64);
      v16h v1 = load_b_frag(&sVt[(wn*32 + 16)*64 + kk*32], 64);
      o0 = WMMA_F32_F16(pf, v0, o0);
      o1 = WMMA_F32_F16(pf, v1, o1);
    }
  }
  __syncthreads();
  #pragma unroll
  for (int r = 0; r < 8; ++r) {
    int m = wm*16 + lh*8 + r;
    float inv = 1.0f / sL[m];
    size_t base = ((size_t)(b*T_ + i0 + m))*D_ + h*DH_ + wn*32 + lr;
    Out[base]      = o0[r]*inv;
    Out[base + 16] = o1[r]*inv;
  }
}

// ---------------------------------------------------------------------------
// Residual add + LayerNorm (biased variance), out f32 + f16.
// ---------------------------------------------------------------------------
__global__ __launch_bounds__(256) void k_resln(const float* __restrict__ Hin,
                                               const float* __restrict__ X,
                                               const float* __restrict__ g,
                                               const float* __restrict__ bb,
                                               float* __restrict__ Hout,
                                               _Float16* __restrict__ Hh)
{
  __shared__ float r1[256], r2[256];
  const int t = threadIdx.x;
  const size_t row = blockIdx.x;
  float x[4]; float s = 0.0f, s2 = 0.0f;
  #pragma unroll
  for (int i = 0; i < 4; ++i) {
    int c = t + i*256;
    float v = Hin[row*D_ + c] + X[row*D_ + c];
    x[i] = v; s += v; s2 += v*v;
  }
  r1[t] = s; r2[t] = s2; __syncthreads();
  for (int off = 128; off > 0; off >>= 1) {
    if (t < off) { r1[t] += r1[t+off]; r2[t] += r2[t+off]; }
    __syncthreads();
  }
  float mean = r1[0] * (1.0f/D_);
  float var  = r2[0] * (1.0f/D_) - mean*mean;
  float inv  = rsqrtf(var + 1e-5f);
  #pragma unroll
  for (int i = 0; i < 4; ++i) {
    int c = t + i*256;
    float y = (x[i] - mean)*inv*g[c] + bb[c];
    Hout[row*D_ + c] = y;
    Hh[row*D_ + c]   = (_Float16)y;
  }
}

// ---------------------------------------------------------------------------
__global__ __launch_bounds__(256) void k_embed(const int* __restrict__ tok,
                                               const float* __restrict__ emb,
                                               const float* __restrict__ pe,
                                               float* __restrict__ H,
                                               _Float16* __restrict__ Hh)
{
  const size_t row = blockIdx.x;
  const int tpos = (int)(row % T_);
  const int id = tok[row];
  const int t = threadIdx.x;
  #pragma unroll
  for (int i = 0; i < 4; ++i) {
    int c = t + i*256;
    float v = emb[(size_t)id*D_ + c] + pe[(size_t)tpos*D_ + c];
    H[row*D_ + c]  = v;
    Hh[row*D_ + c] = (_Float16)v;
  }
}

__global__ void k_cvt(const float* __restrict__ in, _Float16* __restrict__ out,
                      size_t n)
{
  size_t i  = (size_t)blockIdx.x*blockDim.x + threadIdx.x;
  size_t st = (size_t)gridDim.x*blockDim.x;
  for (; i < n; i += st) out[i] = (_Float16)in[i];
}

// ---------------------------------------------------------------------------
extern "C" void kernel_launch(void* const* d_in, const int* in_sizes, int n_in,
                              void* d_out, int out_size, void* d_ws, size_t ws_size,
                              hipStream_t stream)
{
  (void)in_sizes; (void)n_in; (void)out_size; (void)ws_size;
  const int*   tokens = (const int*)  d_in[0];
  const float* embedW = (const float*)d_in[1];
  const float* PE     = (const float*)d_in[2];
  const float* Wq     = (const float*)d_in[3];
  const float* Wk     = (const float*)d_in[4];
  const float* Wv     = (const float*)d_in[5];
  const float* ln1g   = (const float*)d_in[6];
  const float* ln1b   = (const float*)d_in[7];
  const float* W1     = (const float*)d_in[8];
  const float* b1     = (const float*)d_in[9];
  const float* W2     = (const float*)d_in[10];
  const float* b2     = (const float*)d_in[11];
  const float* ln2g   = (const float*)d_in[12];
  const float* ln2b   = (const float*)d_in[13];
  const float* Wlm    = (const float*)d_in[14];
  const float* blm    = (const float*)d_in[15];

  char* ws = (char*)d_ws; size_t off = 0;
  auto carve = [&](size_t bytes) -> void* {
    void* p = ws + off;
    off += (bytes + 255) & ~(size_t)255;
    return p;
  };
  float*    H    = (float*)   carve((size_t)M_*D_*4);
  _Float16* Hh   = (_Float16*)carve((size_t)M_*D_*2);
  _Float16* Qh   = (_Float16*)carve((size_t)M_*D_*2);
  _Float16* Kh   = (_Float16*)carve((size_t)M_*D_*2);
  _Float16* Vh   = (_Float16*)carve((size_t)M_*D_*2);
  float*    Xf   = (float*)   carve((size_t)M_*D_*4);      // attn / MLP2 out
  _Float16* F1h  = (_Float16*)carve((size_t)M_*4*D_*2);
  _Float16* WqH  = (_Float16*)carve((size_t)L_*D_*D_*2);
  _Float16* WkH  = (_Float16*)carve((size_t)L_*D_*D_*2);
  _Float16* WvH  = (_Float16*)carve((size_t)L_*D_*D_*2);
  _Float16* W1H  = (_Float16*)carve((size_t)L_*D_*4*D_*2);
  _Float16* W2H  = (_Float16*)carve((size_t)L_*4*D_*D_*2);
  _Float16* WlmH = (_Float16*)carve((size_t)D_*V_*2);

  // One-time fp32 -> f16 weight conversion (into workspace).
  k_cvt<<<2048, 256, 0, stream>>>(Wq,  WqH,  (size_t)L_*D_*D_);
  k_cvt<<<2048, 256, 0, stream>>>(Wk,  WkH,  (size_t)L_*D_*D_);
  k_cvt<<<2048, 256, 0, stream>>>(Wv,  WvH,  (size_t)L_*D_*D_);
  k_cvt<<<4096, 256, 0, stream>>>(W1,  W1H,  (size_t)L_*D_*4*D_);
  k_cvt<<<4096, 256, 0, stream>>>(W2,  W2H,  (size_t)L_*4*D_*D_);
  k_cvt<<<4096, 256, 0, stream>>>(Wlm, WlmH, (size_t)D_*V_);

  k_embed<<<M_, 256, 0, stream>>>(tokens, embedW, PE, H, Hh);

  for (int l = 0; l < L_; ++l) {
    dim3 gqkv(D_/64, M_/128);
    k_gemm<<<gqkv, 256, 0, stream>>>(Hh, WqH + (size_t)l*D_*D_, nullptr,
                                     nullptr, Qh, M_, D_, D_, 0);
    k_gemm<<<gqkv, 256, 0, stream>>>(Hh, WkH + (size_t)l*D_*D_, nullptr,
                                     nullptr, Kh, M_, D_, D_, 0);
    k_gemm<<<gqkv, 256, 0, stream>>>(Hh, WvH + (size_t)l*D_*D_, nullptr,
                                     nullptr, Vh, M_, D_, D_, 0);
    dim3 ga(T_/64, B_*NH_);
    k_attn<<<ga, 256, 0, stream>>>(Qh, Kh, Vh, Xf);
    k_resln<<<M_, 256, 0, stream>>>(H, Xf, ln1g + (size_t)l*D_,
                                    ln1b + (size_t)l*D_, H, Hh);
    dim3 g1(4*D_/64, M_/128);
    k_gemm<<<g1, 256, 0, stream>>>(Hh, W1H + (size_t)l*D_*4*D_,
                                   b1 + (size_t)l*4*D_, nullptr, F1h,
                                   M_, 4*D_, D_, FLAG_BIAS | FLAG_SELU);
    dim3 g2(D_/64, M_/128);
    k_gemm<<<g2, 256, 0, stream>>>(F1h, W2H + (size_t)l*4*D_*D_,
                                   b2 + (size_t)l*D_, Xf, nullptr,
                                   M_, D_, 4*D_, FLAG_BIAS | FLAG_OUTF32);
    k_resln<<<M_, 256, 0, stream>>>(H, Xf, ln2g + (size_t)l*D_,
                                    ln2b + (size_t)l*D_, H, Hh);
  }
  dim3 gl(V_/64, M_/128);
  k_gemm<<<gl, 256, 0, stream>>>(Hh, WlmH, blm, (float*)d_out, nullptr,
                                 M_, V_, D_, FLAG_BIAS | FLAG_OUTF32);
}
